// VAEPieceDecoder_66434554134894
// MI455X (gfx1250) — matrix operand order repaired and checked by reading, not compiled, not run on hardware
//
#include <hip/hip_runtime.h>
#include <stdint.h>

// ---------------- problem constants ----------------
#define B_     32
#define N_     128
#define L_     48
#define DNODE  160
#define H_     128
#define Z_     64
#define DP_    64
#define HR_    256
#define V_     300
#define NE_    5
#define COND_  256
#define EG_    32768
#define ESEL_  262144
#define TSTEPS 4
#define MLPIN  320
#define MLPHID 160

typedef __attribute__((ext_vector_type(16))) __bf16 v16bf;
typedef __attribute__((ext_vector_type(8)))  float  v8f;
typedef __attribute__((ext_vector_type(4)))  float  f32x4;

// ---------------- WMMA helpers ----------------
__device__ __forceinline__ v8f wmma_bf16(v16bf a, v16bf b, v8f c) {
    return __builtin_amdgcn_wmma_f32_16x16x32_bf16(false, a, false, b, (short)0, c, false, false);
}

__device__ __forceinline__ v8f zero8() {
    v8f v = {};
    return v;
}

// A-matrix fragment (16x32 bf16, M x K) from row-major LDS tile [16][ldk].
// Per ISA layout: per-lane two contiguous 16B chunks at k0 = kc*32 + grp*8 and k0+16.
__device__ __forceinline__ v16bf lds_a_frag(const __bf16* base, int ldk, int kc, int lane) {
    int m   = lane & 15;
    int grp = (lane >> 4) & 1;
    const __bf16* p = base + m * ldk + kc * 32 + grp * 8;
    union { f32x4 f[2]; v16bf v; } u;
    u.f[0] = *(const f32x4*)p;
    u.f[1] = *(const f32x4*)(p + 16);
    return u.v;
}

// B-matrix fragment (32x16 bf16, K x N) from pre-packed global blobs:
// blob layout [blob][lane][16 bf16] -> one contiguous 32B load per lane.
__device__ __forceinline__ v16bf glb_b_frag(const __bf16* packed, int blob, int lane) {
    const v16bf* p = (const v16bf*)(packed + (size_t)blob * 512);
    return p[lane];
}

__device__ __forceinline__ float sigm(float x) { return 1.0f / (1.0f + expf(-x)); }

// ---------------- weight repack: W[K][N] f32 -> B-fragment blobs (bf16) ----------------
__global__ void k_pack(const float* __restrict__ W, __bf16* __restrict__ out,
                       int K, int N, int NT, int KC) {
    int t = blockIdx.x * blockDim.x + threadIdx.x;
    int total = NT * KC * 32;
    if (t >= total) return;
    int lane = t & 31;
    int blob = t >> 5;
    int kc = blob % KC;
    int nt = blob / KC;
    int n = nt * 16 + (lane & 15);
    int kbase = kc * 32 + ((lane >> 4) & 1) * 8;
#pragma unroll
    for (int e = 0; e < 16; ++e) {
        int k = kbase + (e & 7) + ((e >> 3) << 4);
        float v = (k < K && n < N) ? W[(size_t)k * N + n] : 0.0f;
        out[(size_t)blob * 512 + lane * 16 + e] = (__bf16)v;
    }
}

// ---------------- rsample + KL ----------------
__global__ void k_rsample(const float* __restrict__ conds, const float* __restrict__ eps,
                          const float* __restrict__ Wm, const float* __restrict__ bm,
                          const float* __restrict__ Wlv, const float* __restrict__ blv,
                          float* __restrict__ z_out, float* __restrict__ z32,
                          __bf16* __restrict__ zbf, float* __restrict__ accum) {
    int tid = blockIdx.x * 256 + threadIdx.x;   // 2048 threads = B_*Z_
    int b = tid >> 6, d = tid & 63;
    float zm = bm[d], lv = blv[d];
    const float* crow = conds + b * COND_;
    for (int k = 0; k < COND_; ++k) {
        float c = crow[k];
        zm += c * Wm[k * Z_ + d];
        lv += c * Wlv[k * Z_ + d];
    }
    lv = -fabsf(lv);
    float z = zm + expf(0.5f * lv) * eps[tid];
    z_out[tid] = z;
    z32[tid]   = z;
    zbf[tid]   = (__bf16)z;
    float klc = -0.5f * (1.0f + lv - zm * zm - expf(lv));
    __shared__ float sr[256];
    sr[threadIdx.x] = klc;
    __syncthreads();
    for (int s = 128; s > 0; s >>= 1) {
        if (threadIdx.x < s) sr[threadIdx.x] += sr[threadIdx.x + s];
        __syncthreads();
    }
    if (threadIdx.x == 0) atomicAdd(&accum[0], sr[0]);
}

// ---------------- h0 = z @ W_l2h + b ----------------
__global__ void k_h0(const float* __restrict__ z32, const float* __restrict__ W,
                     const float* __restrict__ bias, float* __restrict__ h0) {
    int tid = blockIdx.x * 256 + threadIdx.x;   // 8192
    int b = tid >> 8, c = tid & 255;
    float acc = bias[c];
    const float* zr = z32 + b * Z_;
    for (int k = 0; k < Z_; ++k) acc += zr[k] * W[k * HR_ + c];
    h0[tid] = acc;
}

// ---------------- h = relu(x @ W_in + b_in), WMMA ----------------
__global__ void k_dense_in(const float* __restrict__ x, const __bf16* __restrict__ Bp,
                           const float* __restrict__ bias, float* __restrict__ hout) {
    __shared__ __align__(16) __bf16 sA[8][16][DNODE];
    int wave = threadIdx.x >> 5, lane = threadIdx.x & 31;
    int col16 = lane & 15;
    int mtile = blockIdx.x * 8 + wave;          // 256 tiles total
    int rbase = mtile * 16;
    for (int t = lane; t < 16 * DNODE; t += 32) {
        int m = t / DNODE, c = t % DNODE;
        sA[wave][m][c] = (__bf16)x[(size_t)(rbase + m) * DNODE + c];
    }
    __syncthreads();
    const __bf16* abase = &sA[wave][0][0];
    v8f acc[8];
#pragma unroll
    for (int q = 0; q < 8; ++q) acc[q] = zero8();
    for (int kc = 0; kc < 5; ++kc) {
        v16bf a = lds_a_frag(abase, DNODE, kc, lane);
#pragma unroll
        for (int q = 0; q < 8; ++q)
            acc[q] = wmma_bf16(a, glb_b_frag(Bp, q * 5 + kc, lane), acc[q]);
    }
#pragma unroll
    for (int q = 0; q < 8; ++q) {
        int col = q * 16 + col16;
        float bb = bias[col];
#pragma unroll
        for (int r = 0; r < 8; ++r) {
            int row = rbase + r + ((lane >> 4) << 3);
            hout[(size_t)row * H_ + col] = fmaxf(acc[q][r] + bb, 0.0f);
        }
    }
}

// ---------------- GNN scatter: msg[dst] += relu(h[src] + e_emb) ----------------
__global__ void k_scatter(const float* __restrict__ h, const float* __restrict__ edge_attr,
                          const int* __restrict__ ei, const float* __restrict__ W_edge,
                          const float* __restrict__ b_edge, float* __restrict__ msg) {
    int tid = blockIdx.x * 256 + threadIdx.x;   // EG_*32
    int e = tid >> 5, c4 = tid & 31;
    int src = ei[e], dst = ei[EG_ + e];
    const float* attr = edge_attr + (size_t)e * NE_;
    f32x4 hv = *(const f32x4*)(h + (size_t)src * H_ + c4 * 4);
#pragma unroll
    for (int ii = 0; ii < 4; ++ii) {
        int c = c4 * 4 + ii;
        float emb = b_edge[c];
#pragma unroll
        for (int k = 0; k < NE_; ++k) emb += attr[k] * W_edge[k * H_ + c];
        float v = fmaxf(hv[ii] + emb, 0.0f);
        atomicAdd(&msg[(size_t)dst * H_ + c], v);
    }
}

// ---------------- GNN matmul: hout = relu((h+msg) @ Wg[t] + bg[t]), WMMA ----------------
__global__ void k_gnn_mm(const float* __restrict__ h, const float* __restrict__ msg,
                         const __bf16* __restrict__ Bp, const float* __restrict__ bias,
                         float* __restrict__ hout) {
    __shared__ __align__(16) __bf16 sA[8][16][H_];
    int wave = threadIdx.x >> 5, lane = threadIdx.x & 31;
    int col16 = lane & 15;
    int mtile = blockIdx.x * 8 + wave;
    int rbase = mtile * 16;
    for (int t = lane; t < 16 * H_; t += 32) {
        int m = t / H_, c = t % H_;
        size_t idx = (size_t)(rbase + m) * H_ + c;
        sA[wave][m][c] = (__bf16)(h[idx] + msg[idx]);
    }
    __syncthreads();
    const __bf16* abase = &sA[wave][0][0];
    v8f acc[8];
#pragma unroll
    for (int q = 0; q < 8; ++q) acc[q] = zero8();
    for (int kc = 0; kc < 4; ++kc) {
        v16bf a = lds_a_frag(abase, H_, kc, lane);
#pragma unroll
        for (int q = 0; q < 8; ++q)
            acc[q] = wmma_bf16(a, glb_b_frag(Bp, q * 4 + kc, lane), acc[q]);
    }
#pragma unroll
    for (int q = 0; q < 8; ++q) {
        int col = q * 16 + col16;
        float bb = bias[col];
#pragma unroll
        for (int r = 0; r < 8; ++r) {
            int row = rbase + r + ((lane >> 4) << 3);
            hout[(size_t)row * H_ + col] = fmaxf(acc[q][r] + bb, 0.0f);
        }
    }
}

__global__ void k_tobf(const float* __restrict__ in, __bf16* __restrict__ out) {
    int tid = blockIdx.x * 256 + threadIdx.x;   // 524288
    out[tid] = (__bf16)in[tid];
}

// ---------------- Edge-pair MLP (the hot kernel, ~68 GFLOP) ----------------
// 2 waves / block; each wave owns a 16-row tile: ein[16x320] -> 160 -> 160 -> 320 -> 5
__global__ void k_edge_mlp(const __bf16* __restrict__ node_bf, const __bf16* __restrict__ zbf,
                           const int* __restrict__ sel_b, const int* __restrict__ sel_i,
                           const int* __restrict__ sel_j, const int* __restrict__ golden,
                           const __bf16* __restrict__ W1p, const __bf16* __restrict__ W2p,
                           const __bf16* __restrict__ W3p, const __bf16* __restrict__ W4p,
                           const float* __restrict__ b1, const float* __restrict__ b2,
                           const float* __restrict__ b3, const float* __restrict__ b4,
                           float* __restrict__ accum) {
    __shared__ __align__(16) __bf16 sA[2][16][MLPIN];    // ein, later h3
    __shared__ __align__(16) __bf16 sH1[2][16][MLPHID];
    __shared__ __align__(16) __bf16 sH2[2][16][MLPHID];
    __shared__ float sLog[2][16][16];
    __shared__ float sRed[64];

    int wave = threadIdx.x >> 5, lane = threadIdx.x & 31;
    int col16 = lane & 15;
    int mrow  = (lane >> 4) << 3;
    float nll_acc = 0.0f;

    for (int it = 0; it < 4; ++it) {
        int tile = (blockIdx.x * 2 + wave) * 4 + it;
        // prefetch next tile's selection indices (random-access stream)
        {
            int nxt = (tile + 1) * 16;
            __builtin_prefetch(&sel_b[nxt], 0, 0);
            __builtin_prefetch(&sel_i[nxt], 0, 0);
            __builtin_prefetch(&sel_j[nxt], 0, 0);
            __builtin_prefetch(&golden[nxt], 0, 0);
        }
        // ---- gather ein (src_e | dst_e | lat) as bf16 dwords ----
        for (int m = 0; m < 16; ++m) {
            int idx = tile * 16 + m;
            int bb = sel_b[idx], ii = sel_i[idx], jj = sel_j[idx];
            const uint32_t* srcp = (const uint32_t*)(node_bf + ((size_t)bb * N_ + ii) * H_);
            const uint32_t* dstp = (const uint32_t*)(node_bf + ((size_t)bb * N_ + jj) * H_);
            const uint32_t* latp = (const uint32_t*)(zbf + (size_t)bb * Z_);
            uint32_t* arow = (uint32_t*)&sA[wave][m][0];
            for (int d = lane; d < 160; d += 32) {
                uint32_t u = (d < 64) ? srcp[d] : (d < 128 ? dstp[d - 64] : latp[d - 128]);
                arow[d] = u;
            }
        }
        __syncthreads();
        // ---- layer 1: [16x320] @ W1 -> relu -> sH1 ----
        {
            v8f acc[10];
#pragma unroll
            for (int q = 0; q < 10; ++q) acc[q] = zero8();
            for (int kc = 0; kc < 10; ++kc) {
                v16bf a = lds_a_frag(&sA[wave][0][0], MLPIN, kc, lane);
#pragma unroll
                for (int q = 0; q < 10; ++q)
                    acc[q] = wmma_bf16(a, glb_b_frag(W1p, q * 10 + kc, lane), acc[q]);
            }
#pragma unroll
            for (int q = 0; q < 10; ++q) {
                int col = q * 16 + col16;
                float bb = b1[col];
#pragma unroll
                for (int r = 0; r < 8; ++r)
                    sH1[wave][r + mrow][col] = (__bf16)fmaxf(acc[q][r] + bb, 0.0f);
            }
        }
        __syncthreads();
        // ---- layer 2 ----
        {
            v8f acc[10];
#pragma unroll
            for (int q = 0; q < 10; ++q) acc[q] = zero8();
            for (int kc = 0; kc < 5; ++kc) {
                v16bf a = lds_a_frag(&sH1[wave][0][0], MLPHID, kc, lane);
#pragma unroll
                for (int q = 0; q < 10; ++q)
                    acc[q] = wmma_bf16(a, glb_b_frag(W2p, q * 5 + kc, lane), acc[q]);
            }
#pragma unroll
            for (int q = 0; q < 10; ++q) {
                int col = q * 16 + col16;
                float bb = b2[col];
#pragma unroll
                for (int r = 0; r < 8; ++r)
                    sH2[wave][r + mrow][col] = (__bf16)fmaxf(acc[q][r] + bb, 0.0f);
            }
        }
        __syncthreads();
        // ---- layer 3 (no relu) -> sA reused; two 10-tile halves ----
        for (int hf = 0; hf < 2; ++hf) {
            v8f acc[10];
#pragma unroll
            for (int q = 0; q < 10; ++q) acc[q] = zero8();
            for (int kc = 0; kc < 5; ++kc) {
                v16bf a = lds_a_frag(&sH2[wave][0][0], MLPHID, kc, lane);
#pragma unroll
                for (int q = 0; q < 10; ++q)
                    acc[q] = wmma_bf16(a, glb_b_frag(W3p, (hf * 10 + q) * 5 + kc, lane), acc[q]);
            }
#pragma unroll
            for (int q = 0; q < 10; ++q) {
                int col = (hf * 10 + q) * 16 + col16;
                float bb = b3[col];
#pragma unroll
                for (int r = 0; r < 8; ++r)
                    sA[wave][r + mrow][col] = (__bf16)(acc[q][r] + bb);
            }
        }
        __syncthreads();
        // ---- layer 4: logits (N=5 padded to 16) ----
        {
            v8f acc = zero8();
            for (int kc = 0; kc < 10; ++kc)
                acc = wmma_bf16(lds_a_frag(&sA[wave][0][0], MLPIN, kc, lane),
                                glb_b_frag(W4p, kc, lane), acc);
            float bb = (col16 < NE_) ? b4[col16] : 0.0f;
#pragma unroll
            for (int r = 0; r < 8; ++r)
                sLog[wave][r + mrow][col16] = acc[r] + bb;
        }
        __syncthreads();
        // ---- NLL over the 5 real classes ----
        if (lane < 16) {
            const float* lg = sLog[wave][lane];
            float mx = lg[0];
#pragma unroll
            for (int c = 1; c < NE_; ++c) mx = fmaxf(mx, lg[c]);
            float s = 0.0f;
#pragma unroll
            for (int c = 0; c < NE_; ++c) s += expf(lg[c] - mx);
            int g = golden[tile * 16 + lane];
            nll_acc += (mx + logf(s)) - lg[g];
        }
        __syncthreads();
    }
    sRed[threadIdx.x] = nll_acc;
    __syncthreads();
    if (threadIdx.x == 0) {
        float tot = 0.0f;
        for (int i = 0; i < 64; ++i) tot += sRed[i];
        atomicAdd(&accum[3], tot);
    }
}

// ---------------- GRU language model (single persistent workgroup, WMMA) ----------------
__global__ void k_gru(const int* __restrict__ pieces, const float* __restrict__ piece_emb,
                      const float* __restrict__ h0,
                      const __bf16* __restrict__ Wip, const __bf16* __restrict__ Whp,
                      const __bf16* __restrict__ Wvp,
                      const float* __restrict__ bi, const float* __restrict__ bh,
                      const float* __restrict__ bv, float* __restrict__ accum) {
    __shared__ __align__(16) __bf16 sh[2][B_][HR_];     // 32 KB, ping-pong hidden state
    __shared__ __align__(16) __bf16 sx[B_][DP_];        // 4 KB, current embeddings
    __shared__ __align__(16) __bf16 slog[B_][304];      // 19 KB, vocab logits (padded)

    int tid = threadIdx.x;
    int wave = tid >> 5, lane = tid & 31;
    int col16 = lane & 15;
    int mrow  = (lane >> 4) << 3;

    for (int t = tid; t < B_ * HR_; t += 256)
        sh[0][t / HR_][t % HR_] = (__bf16)h0[t];
    __syncthreads();

    int cur = 0;
    float nll_acc = 0.0f, cnt_acc = 0.0f;

    for (int t = 0; t < L_ - 1; ++t) {
        // embeddings for step t
        for (int q = tid; q < B_ * DP_; q += 256) {
            int b = q >> 6, d = q & 63;
            int pc = pieces[b * L_ + t];
            sx[b][d] = (__bf16)piece_emb[(size_t)pc * DP_ + d];
        }
        __syncthreads();
        // gates: 32 (mtile, ctile) tasks over 8 waves
        for (int task = wave; task < 32; task += 8) {
            int mt = task >> 4, ct = task & 15;
            int col = ct * 16 + col16;
            v8f air = zero8(), aiz = zero8(), ain = zero8();
            v8f ahr = zero8(), ahz = zero8(), ahn = zero8();
            for (int kc = 0; kc < 2; ++kc) {               // gi: K = 64
                v16bf a = lds_a_frag(&sx[mt * 16][0], DP_, kc, lane);
                air = wmma_bf16(a, glb_b_frag(Wip, (ct)      * 2 + kc, lane), air);
                aiz = wmma_bf16(a, glb_b_frag(Wip, (ct + 16) * 2 + kc, lane), aiz);
                ain = wmma_bf16(a, glb_b_frag(Wip, (ct + 32) * 2 + kc, lane), ain);
            }
            for (int kc = 0; kc < 8; ++kc) {               // gh: K = 256
                v16bf a = lds_a_frag(&sh[cur][mt * 16][0], HR_, kc, lane);
                ahr = wmma_bf16(a, glb_b_frag(Whp, (ct)      * 8 + kc, lane), ahr);
                ahz = wmma_bf16(a, glb_b_frag(Whp, (ct + 16) * 8 + kc, lane), ahz);
                ahn = wmma_bf16(a, glb_b_frag(Whp, (ct + 32) * 8 + kc, lane), ahn);
            }
            float bir = bi[col],           bhr = bh[col];
            float biz = bi[HR_ + col],     bhz = bh[HR_ + col];
            float bin = bi[2 * HR_ + col], bhn = bh[2 * HR_ + col];
#pragma unroll
            for (int r = 0; r < 8; ++r) {
                int row = mt * 16 + r + mrow;
                float hp = (float)sh[cur][row][col];
                float rg = sigm(air[r] + bir + ahr[r] + bhr);
                float zg = sigm(aiz[r] + biz + ahz[r] + bhz);
                float ng = tanhf(ain[r] + bin + rg * (ahn[r] + bhn));
                sh[cur ^ 1][row][col] = (__bf16)((1.0f - zg) * ng + zg * hp);
            }
        }
        __syncthreads();
        // vocab logits: out @ W_vocab, 38 (mtile, ntile) tasks
        for (int task = wave; task < 38; task += 8) {
            int mt = task & 1, nt = task >> 1;
            int col = nt * 16 + col16;
            v8f acc = zero8();
            for (int kc = 0; kc < 8; ++kc) {
                v16bf a = lds_a_frag(&sh[cur ^ 1][mt * 16][0], HR_, kc, lane);
                acc = wmma_bf16(a, glb_b_frag(Wvp, nt * 8 + kc, lane), acc);
            }
            float bb = (col < V_) ? bv[col] : 0.0f;
#pragma unroll
            for (int r = 0; r < 8; ++r) {
                int row = mt * 16 + r + mrow;
                slog[row][col] = (col < V_) ? (__bf16)(acc[r] + bb) : (__bf16)(-1e30f);
            }
        }
        __syncthreads();
        // masked NLL for gold piece at position t+1
        if (tid < B_) {
            int b = tid;
            int gold = pieces[b * L_ + t + 1];
            float mx = -1e30f;
            for (int c = 0; c < V_; ++c) mx = fmaxf(mx, (float)slog[b][c]);
            float s = 0.0f;
            for (int c = 0; c < V_; ++c) s += expf((float)slog[b][c] - mx);
            float nll = mx + logf(s) - (float)slog[b][gold];
            if (gold != 0) { nll_acc += nll; cnt_acc += 1.0f; }
        }
        __syncthreads();
        cur ^= 1;
    }
    if (tid < B_) {
        atomicAdd(&accum[1], nll_acc);
        atomicAdd(&accum[2], cnt_acc);
    }
}

// ---------------- finalize losses ----------------
__global__ void k_finalize(const float* __restrict__ accum, float* __restrict__ out) {
    float kl = accum[0] / (float)B_;
    float pl = accum[1] / accum[2];
    float el = accum[3] / (float)ESEL_;
    out[2048] = pl + el;
    out[2049] = pl;
    out[2050] = el;
    out[2051] = kl;
}

// ---------------- host launch ----------------
extern "C" void kernel_launch(void* const* d_in, const int* in_sizes, int n_in,
                              void* d_out, int out_size, void* d_ws, size_t ws_size,
                              hipStream_t stream) {
    // inputs (setup_inputs order, params flattened in insertion order)
    const float* x          = (const float*)d_in[0];
    const float* edge_attr  = (const float*)d_in[1];
    const float* conds      = (const float*)d_in[2];
    const float* epsilon    = (const float*)d_in[3];
    const int*   pieces     = (const int*)d_in[4];
    const int*   edge_index = (const int*)d_in[5];
    const int*   sel_b      = (const int*)d_in[6];
    const int*   sel_i      = (const int*)d_in[7];
    const int*   sel_j      = (const int*)d_in[8];
    const int*   golden     = (const int*)d_in[9];
    const float* W_mean = (const float*)d_in[10]; const float* b_mean = (const float*)d_in[11];
    const float* W_lv   = (const float*)d_in[12]; const float* b_lv   = (const float*)d_in[13];
    const float* piece_emb = (const float*)d_in[14];
    const float* W_l2h  = (const float*)d_in[15]; const float* b_l2h  = (const float*)d_in[16];
    const float* Wi     = (const float*)d_in[17]; const float* bi     = (const float*)d_in[18];
    const float* Wh     = (const float*)d_in[19]; const float* bh     = (const float*)d_in[20];
    const float* W_voc  = (const float*)d_in[21]; const float* b_voc  = (const float*)d_in[22];
    const float* W_in   = (const float*)d_in[23]; const float* b_in   = (const float*)d_in[24];
    const float* W_edge = (const float*)d_in[25]; const float* b_edge = (const float*)d_in[26];
    const float* W_gnn  = (const float*)d_in[27]; const float* b_gnn  = (const float*)d_in[28];
    const float* W1 = (const float*)d_in[29]; const float* b1 = (const float*)d_in[30];
    const float* W2 = (const float*)d_in[31]; const float* b2 = (const float*)d_in[32];
    const float* W3 = (const float*)d_in[33]; const float* b3 = (const float*)d_in[34];
    const float* W4 = (const float*)d_in[35]; const float* b4 = (const float*)d_in[36];
    float* out = (float*)d_out;

    // workspace layout
    char* cur = (char*)d_ws;
    auto alloc = [&](size_t bytes) -> void* {
        void* p = (void*)cur;
        cur += (bytes + 255) & ~(size_t)255;
        return p;
    };
    float*  accum   = (float*)alloc(4 * sizeof(float));
    float*  z32     = (float*)alloc(B_ * Z_ * 4);
    __bf16* zbf     = (__bf16*)alloc(B_ * Z_ * 2);
    float*  h0      = (float*)alloc(B_ * HR_ * 4);
    float*  hA      = (float*)alloc((size_t)B_ * N_ * H_ * 4);
    float*  hB      = (float*)alloc((size_t)B_ * N_ * H_ * 4);
    float*  msg     = (float*)alloc((size_t)B_ * N_ * H_ * 4);
    __bf16* node_bf = (__bf16*)alloc((size_t)B_ * N_ * H_ * 2);
    __bf16* W1p  = (__bf16*)alloc((size_t)10 * 10 * 512 * 2);
    __bf16* W2p  = (__bf16*)alloc((size_t)10 * 5 * 512 * 2);
    __bf16* W3p  = (__bf16*)alloc((size_t)20 * 5 * 512 * 2);
    __bf16* W4p  = (__bf16*)alloc((size_t)1 * 10 * 512 * 2);
    __bf16* Wip  = (__bf16*)alloc((size_t)48 * 2 * 512 * 2);
    __bf16* Whp  = (__bf16*)alloc((size_t)48 * 8 * 512 * 2);
    __bf16* Wvp  = (__bf16*)alloc((size_t)19 * 8 * 512 * 2);
    __bf16* Winp = (__bf16*)alloc((size_t)8 * 5 * 512 * 2);
    __bf16* Wgp  = (__bf16*)alloc((size_t)TSTEPS * 8 * 4 * 512 * 2);

    hipMemsetAsync(accum, 0, 4 * sizeof(float), stream);

    auto pack = [&](const float* W, __bf16* outp, int K, int N, int NT, int KC) {
        int total = NT * KC * 32;
        k_pack<<<(total + 255) / 256, 256, 0, stream>>>(W, outp, K, N, NT, KC);
    };
    pack(W1, W1p, MLPIN, MLPHID, 10, 10);
    pack(W2, W2p, MLPHID, MLPHID, 10, 5);
    pack(W3, W3p, MLPHID, MLPIN, 20, 5);
    pack(W4, W4p, MLPIN, NE_, 1, 10);
    pack(Wi, Wip, DP_, 3 * HR_, 48, 2);
    pack(Wh, Whp, HR_, 3 * HR_, 48, 8);
    pack(W_voc, Wvp, HR_, V_, 19, 8);
    pack(W_in, Winp, DNODE, H_, 8, 5);
    for (int t = 0; t < TSTEPS; ++t)
        pack(W_gnn + (size_t)t * H_ * H_, Wgp + (size_t)t * 8 * 4 * 512, H_, H_, 8, 4);

    // rsample + h0
    k_rsample<<<8, 256, 0, stream>>>(conds, epsilon, W_mean, b_mean, W_lv, b_lv,
                                     out, z32, zbf, accum);
    k_h0<<<32, 256, 0, stream>>>(z32, W_l2h, b_l2h, h0);

    // graph encoder
    k_dense_in<<<32, 256, 0, stream>>>(x, Winp, b_in, hA);
    float* hin = hA;
    float* hout = hB;
    for (int t = 0; t < TSTEPS; ++t) {
        hipMemsetAsync(msg, 0, (size_t)B_ * N_ * H_ * 4, stream);
        k_scatter<<<4096, 256, 0, stream>>>(hin, edge_attr, edge_index, W_edge, b_edge, msg);
        k_gnn_mm<<<32, 256, 0, stream>>>(hin, msg, Wgp + (size_t)t * 8 * 4 * 512,
                                         b_gnn + t * H_, hout);
        float* tmp = hin; hin = hout; hout = tmp;
    }
    k_tobf<<<2048, 256, 0, stream>>>(hin, node_bf);

    // edge-pair MLP (dominant compute)
    k_edge_mlp<<<2048, 64, 0, stream>>>(node_bf, zbf, sel_b, sel_i, sel_j, golden,
                                        W1p, W2p, W3p, W4p, b1, b2, b3, b4, accum);

    // GRU LM
    k_gru<<<1, 256, 0, stream>>>(pieces, piece_emb, h0, Wip, Whp, Wvp, bi, bh, b_voc, accum);

    // losses
    k_finalize<<<1, 1, 0, stream>>>(accum, out);
}
